// CNNMambaClassifier_90735479095333
// MI455X (gfx1250) — matrix operand order repaired and checked by reading, not compiled, not run on hardware
//
#include <hip/hip_runtime.h>
#include <hip/hip_bf16.h>

// ---------------------------------------------------------------------------
// CDNA5 (gfx1250, wave32) ResNet50 + Mamba classifier.
// Conv / projection GEMMs use V_WMMA_F32_16X16X32_BF16 with double-buffered
// LDS tiles filled by GLOBAL_LOAD_ASYNC_TO_LDS_B128 (ASYNCcnt pipelining).
// ---------------------------------------------------------------------------

typedef __attribute__((ext_vector_type(16))) __bf16 v16bf;
typedef __attribute__((ext_vector_type(8)))  float  v8f;
typedef __attribute__((ext_vector_type(4)))  int    v4i;

#define TB 256  // 8 waves of 32

#if __has_builtin(__builtin_amdgcn_global_load_async_to_lds_b128) && \
    __has_builtin(__builtin_amdgcn_s_wait_asynccnt)
#define USE_ASYNC_LDS 1
#define AS1 __attribute__((address_space(1)))
#define AS3 __attribute__((address_space(3)))
__device__ __forceinline__ void async_b128(const void* g, void* l) {
    __builtin_amdgcn_global_load_async_to_lds_b128((AS1 v4i*)g, (AS3 v4i*)l, 0, 0);
}
#define ASYNC_WAIT(n) __builtin_amdgcn_s_wait_asynccnt(n)
#else
#define USE_ASYNC_LDS 0
#define ASYNC_WAIT(n)
#endif

// LDS tile row stride: 40 bf16 = 80 B (16B-aligned chunks; 20-dword stride is
// conflict-free for the 16-lane fragment read pattern).
#define LDSP 40

// ---------------------------------------------------------------------------
// fp32 -> bf16 conversion (weights / activations)
// ---------------------------------------------------------------------------
__global__ void cvt_f32_bf16(const float* __restrict__ in, __bf16* __restrict__ out, int n) {
    int i = blockIdx.x * blockDim.x + threadIdx.x;
    if (i < n) out[i] = (__bf16)in[i];
}

// ---------------------------------------------------------------------------
// Stem: 7x7 s2 p3 conv 3->64 + BN + ReLU, fp32 in, bf16 out (C=3 -> VALU)
// ---------------------------------------------------------------------------
__global__ void stem_conv(const float* __restrict__ x, __bf16* __restrict__ out,
                          const float* __restrict__ w, const float* __restrict__ sc,
                          const float* __restrict__ bi) {
    int idx = blockIdx.x * blockDim.x + threadIdx.x;
    const int TOT = 16 * 64 * 112 * 112;
    if (idx >= TOT) return;
    int ox = idx % 112; int t = idx / 112;
    int oy = t % 112; t /= 112;
    int k  = t % 64;  int n = t / 64;
    float acc = 0.f;
    for (int c = 0; c < 3; ++c)
        for (int r = 0; r < 7; ++r) {
            int iy = oy * 2 + r - 3;
            if (iy < 0 || iy >= 224) continue;
            for (int s = 0; s < 7; ++s) {
                int ix = ox * 2 + s - 3;
                if (ix < 0 || ix >= 224) continue;
                acc += x[((n * 3 + c) * 224 + iy) * 224 + ix] *
                       w[((k * 3 + c) * 7 + r) * 7 + s];
            }
        }
    float v = acc * sc[k] + bi[k];
    out[idx] = (__bf16)(v > 0.f ? v : 0.f);
}

// ---------------------------------------------------------------------------
// 3x3 s2 p1 maxpool, bf16
// ---------------------------------------------------------------------------
__global__ void maxpool3(const __bf16* __restrict__ in, __bf16* __restrict__ out) {
    int idx = blockIdx.x * blockDim.x + threadIdx.x;
    const int TOT = 16 * 64 * 56 * 56;
    if (idx >= TOT) return;
    int ox = idx % 56; int t = idx / 56;
    int oy = t % 56;  int ch = t / 56;
    float m = -3.0e38f;
    for (int r = 0; r < 3; ++r) {
        int iy = oy * 2 + r - 1;
        if (iy < 0 || iy >= 112) continue;
        for (int s = 0; s < 3; ++s) {
            int ix = ox * 2 + s - 1;
            if (ix < 0 || ix >= 112) continue;
            float v = (float)in[((size_t)ch * 112 + iy) * 112 + ix];
            m = v > m ? v : m;
        }
    }
    out[idx] = (__bf16)m;
}

// ---------------------------------------------------------------------------
// Implicit-GEMM conv (WMMA bf16), software-pipelined double-buffered LDS:
//   GEMM: M = out-ch, N = B*Hout*Wout, Kdim = Cin*R*S (multiple of 32).
// Macro-tile 128x128x32, 8 waves in 4(M)x2(N) grid, 8 v_wmma per wave per
// K-step. Tile t+1 is staged (A: async b128; B: im2col gather held in regs)
// while tile t runs on the matrix pipe; s_wait_asynccnt(1) drains tile t only.
// Each thread owns ONE pixel (n = tid&127) -> pixel decode hoisted out of the
// K-loop; only c = kd/9 remains (literal divisor -> mul-shift).
// ---------------------------------------------------------------------------
__global__ __launch_bounds__(TB) void conv_gemm_bf16(
    const __bf16* __restrict__ Wt,   // [M][Kdim]
    const __bf16* __restrict__ In,   // [B][Cin][Hin][Win]
    __bf16* __restrict__ Out,        // [B][M][Hout][Wout]
    const float* __restrict__ scale, const float* __restrict__ bias,
    const __bf16* __restrict__ resid, int relu,
    int M, int Cin, int Hin, int Win, int Hout, int Wout,
    int RS1,                          // 1 if 1x1 conv, 0 if 3x3
    int stride, int pad, int Batch)
{
    const int HWo = Hout * Wout, HWi = Hin * Win;
    const int Kdim = RS1 ? Cin : Cin * 9;
    const int N = Batch * HWo;
    const int gn0 = blockIdx.x * 128;
    const int gm0 = blockIdx.y * 128;
    const int tid = threadIdx.x;
    const int lane = tid & 31, wave = tid >> 5;
    const int mwq = wave & 3, nw = wave >> 2;     // 4(M) x 2(N) wave grid
    const int fm = lane & 15, hi = lane >> 4;

    __shared__ __bf16 As[2][128][LDSP];
    __shared__ __bf16 Bs[2][128][LDSP];

    v8f acc[2][4] = {};

    // ---- per-thread loop invariants for the B (im2col) gather ----
    const int n    = tid & 127;       // this thread's pixel slot
    const int kpar = tid >> 7;        // K-slot parity (kk = kpar + 2t)
    const int gn   = gn0 + n;
    const bool gnok = gn < N;
    const int pb  = gn / HWo, pyx = gn % HWo;
    const int poy = pyx / Wout, pox = pyx % Wout;
    const int ibase = pb * Cin * HWi;
    const int iy0 = poy * stride - pad, ix0 = pox * stride - pad;

    // gather one 32-K slab of the im2col matrix into registers
    auto gatherB = [&](int k0, __bf16* r) {
#pragma unroll
        for (int t = 0; t < 16; ++t) {
            int kd = k0 + kpar + 2 * t;
            int c, iy, ix;
            if (RS1) { c = kd; iy = poy * stride; ix = pox * stride; }
            else {
                c = kd / 9; int rs = kd % 9;            // literal 9 -> mul/shift
                iy = iy0 + rs / 3; ix = ix0 + rs % 3;
            }
            bool ok = gnok && iy >= 0 && iy < Hin && ix >= 0 && ix < Win;
            int a = ok ? (ibase + c * HWi + iy * Win + ix) : 0;
            __bf16 raw = In[a];                         // always in-bounds
            r[t] = ok ? raw : (__bf16)0.f;
        }
    };
    auto storeB = [&](int buf, const __bf16* r) {
#pragma unroll
        for (int t = 0; t < 16; ++t) Bs[buf][n][kpar + 2 * t] = r[t];
    };
    auto stageA = [&](int buf, int k0) {
#if USE_ASYNC_LDS
        int m = tid >> 1, half = tid & 1;               // 128 rows x 2 x 16B
        int gm = gm0 + m;
        if (gm < M)
            async_b128(&Wt[(size_t)gm * Kdim + k0 + 16 * half], &As[buf][m][16 * half]);
        else
            *(uint4*)&As[buf][m][16 * half] = make_uint4(0u, 0u, 0u, 0u);
#else
#pragma unroll
        for (int t = 0; t < 16; ++t) {
            int idx = tid + t * TB;
            int m = idx >> 5, kk = idx & 31;
            int gm = gm0 + m;
            As[buf][m][kk] = (gm < M) ? Wt[(size_t)gm * Kdim + k0 + kk] : (__bf16)0.f;
        }
#endif
    };
    auto compute = [&](int buf) {
        union F { v16bf v; unsigned int u[8]; };
        F af[2];
#pragma unroll
        for (int ai = 0; ai < 2; ++ai) {
            int row = (mwq * 2 + ai) * 16 + fm;
#pragma unroll
            for (int i = 0; i < 8; ++i) {
                int k = 2 * i + (i > 3 ? 8 : 0) + 8 * hi;   // ISA 7.12.2 A layout
                af[ai].u[i] = *(const unsigned int*)&As[buf][row][k];
            }
        }
#pragma unroll
        for (int j = 0; j < 4; ++j) {
            F bfr;
            int row = nw * 64 + j * 16 + fm;
#pragma unroll
            for (int i = 0; i < 8; ++i) {
                int k = 2 * i + (i > 3 ? 8 : 0) + 8 * hi;   // B: same K layout as A
                bfr.u[i] = *(const unsigned int*)&Bs[buf][row][k];
            }
#pragma unroll
            for (int ai = 0; ai < 2; ++ai)
                acc[ai][j] = __builtin_amdgcn_wmma_f32_16x16x32_bf16(
                    false, af[ai].v, false, bfr.v, (short)0, acc[ai][j], false, false);
        }
    };

    // ---- software pipeline over K tiles (buf[t&1] holds tile t) ----
    const int T = Kdim >> 5;
    __bf16 rB[16], rB2[16];
    gatherB(0, rB); storeB(0, rB); stageA(0, 0);
    if (T > 1) gatherB(32, rB2);

    for (int it = 0; it < T; ++it) {
        int cur = it & 1, nxt = cur ^ 1;
        if (it + 1 < T) {
            stageA(nxt, (it + 1) * 32);          // async: overlaps compute below
            storeB(nxt, rB2);                    // regs gathered one tile ago
        }
        if (it + 2 < T) gatherB((it + 2) * 32, rB2);  // loads fly over compute
        if (it + 1 < T) { ASYNC_WAIT(1); }       // drain tile it, keep it+1 inflight
        else            { ASYNC_WAIT(0); }
        __syncthreads();                          // tile `cur` complete for all
        compute(cur);
        __syncthreads();                          // all reads done before overwrite
    }

    // ---- epilogue (D layout: lanes 0-15 -> M=r, 16-31 -> M=8+r; N = lane&15)
#pragma unroll
    for (int j = 0; j < 4; ++j) {
        int nn = gn0 + nw * 64 + j * 16 + fm;
        if (nn >= N) continue;
        int b = nn / HWo, yx = nn % HWo;
#pragma unroll
        for (int ai = 0; ai < 2; ++ai) {
#pragma unroll
            for (int r = 0; r < 8; ++r) {
                int m = gm0 + (mwq * 2 + ai) * 16 + r + 8 * hi;
                if (m >= M) continue;
                float v = acc[ai][j][r];
                float s = scale ? scale[m] : 1.0f;
                float bb = bias ? bias[m] : 0.0f;
                v = v * s + bb;
                size_t oi = ((size_t)b * M + m) * HWo + yx;
                if (resid) v += (float)resid[oi];
                if (relu) v = v > 0.f ? v : 0.f;
                Out[oi] = (__bf16)v;
            }
        }
    }
}

// ---------------------------------------------------------------------------
// Plain WMMA GEMM (token-major): out[n][m] = sum_k W[m][k] * X[n][k] (+bias)
// Fully async double-buffered staging (A: 1 instr, B: 2 instrs per tile).
// ---------------------------------------------------------------------------
__global__ __launch_bounds__(TB) void gemm_bf16(
    const __bf16* __restrict__ Wt,  // [M][K]
    const __bf16* __restrict__ Xr,  // [N][K]
    float* __restrict__ Out,        // [N][M]
    const float* __restrict__ bias,
    int M, int N, int Kdim)
{
    const int gn0 = blockIdx.x * 128;
    const int gm0 = blockIdx.y * 128;
    const int tid = threadIdx.x;
    const int lane = tid & 31, wave = tid >> 5;
    const int mwq = wave & 3, nw = wave >> 2;
    const int fm = lane & 15, hi = lane >> 4;

    __shared__ __bf16 As[2][128][LDSP];
    __shared__ __bf16 Bs[2][128][LDSP];

    v8f acc[2][4] = {};

    auto stage = [&](int buf, int k0) {
#if USE_ASYNC_LDS
        {
            int m = tid >> 1, half = tid & 1;
            int gm = gm0 + m;
            if (gm < M)
                async_b128(&Wt[(size_t)gm * Kdim + k0 + 16 * half], &As[buf][m][16 * half]);
            else
                *(uint4*)&As[buf][m][16 * half] = make_uint4(0u, 0u, 0u, 0u);
        }
#pragma unroll
        for (int t = 0; t < 2; ++t) {             // 128 rows x 4 x 8-elem chunks
            int c = tid + t * TB;
            int nn = c >> 2, q = c & 3;
            int gn = gn0 + nn;
            if (gn < N)
                async_b128(&Xr[(size_t)gn * Kdim + k0 + 8 * q], &Bs[buf][nn][8 * q]);
            else
                *(uint4*)&Bs[buf][nn][8 * q] = make_uint4(0u, 0u, 0u, 0u);
        }
#else
#pragma unroll
        for (int t = 0; t < 16; ++t) {
            int idx = tid + t * TB;
            int m = idx >> 5, kk = idx & 31;
            int gm = gm0 + m;
            As[buf][m][kk] = (gm < M) ? Wt[(size_t)gm * Kdim + k0 + kk] : (__bf16)0.f;
        }
#pragma unroll
        for (int t = 0; t < 16; ++t) {
            int idx = tid + t * TB;
            int kk = idx & 31, nn = idx >> 5;
            int gn = gn0 + nn;
            Bs[buf][nn][kk] = (gn < N) ? Xr[(size_t)gn * Kdim + k0 + kk] : (__bf16)0.f;
        }
#endif
    };
    auto compute = [&](int buf) {
        union F { v16bf v; unsigned int u[8]; };
        F af[2];
#pragma unroll
        for (int ai = 0; ai < 2; ++ai) {
            int row = (mwq * 2 + ai) * 16 + fm;
#pragma unroll
            for (int i = 0; i < 8; ++i) {
                int k = 2 * i + (i > 3 ? 8 : 0) + 8 * hi;
                af[ai].u[i] = *(const unsigned int*)&As[buf][row][k];
            }
        }
#pragma unroll
        for (int j = 0; j < 4; ++j) {
            F bfr;
            int row = nw * 64 + j * 16 + fm;
#pragma unroll
            for (int i = 0; i < 8; ++i) {
                int k = 2 * i + (i > 3 ? 8 : 0) + 8 * hi;
                bfr.u[i] = *(const unsigned int*)&Bs[buf][row][k];
            }
#pragma unroll
            for (int ai = 0; ai < 2; ++ai)
                acc[ai][j] = __builtin_amdgcn_wmma_f32_16x16x32_bf16(
                    false, af[ai].v, false, bfr.v, (short)0, acc[ai][j], false, false);
        }
    };

    const int T = Kdim >> 5;
    stage(0, 0);
    for (int it = 0; it < T; ++it) {
        int cur = it & 1, nxt = cur ^ 1;
        if (it + 1 < T) { stage(nxt, (it + 1) * 32); ASYNC_WAIT(3); }
        else            { ASYNC_WAIT(0); }
        __syncthreads();
        compute(cur);
        __syncthreads();
    }

#pragma unroll
    for (int j = 0; j < 4; ++j) {
        int nn = gn0 + nw * 64 + j * 16 + fm;
        if (nn >= N) continue;
#pragma unroll
        for (int ai = 0; ai < 2; ++ai) {
#pragma unroll
            for (int r = 0; r < 8; ++r) {
                int m = gm0 + (mwq * 2 + ai) * 16 + r + 8 * hi;
                if (m >= M) continue;
                float v = acc[ai][j][r];
                if (bias) v += bias[m];
                Out[(size_t)nn * M + m] = v;
            }
        }
    }
}

// ---------------------------------------------------------------------------
// NCHW bf16 [16,256,49] -> token-major bf16 [784,256]
// ---------------------------------------------------------------------------
__global__ void tokens_from_nchw(const __bf16* __restrict__ src, __bf16* __restrict__ dst) {
    int idx = blockIdx.x * blockDim.x + threadIdx.x;
    if (idx >= 784 * 256) return;
    int d = idx % 256, n = idx / 256;
    int b = n / 49, l = n % 49;
    dst[idx] = src[((size_t)b * 256 + d) * 49 + l];
}

// ---------------------------------------------------------------------------
// Mamba: causal depthwise conv1d (k=4, left pad 3) + bias + SiLU.
// XZ layout [784][1024] (first 512 = x-path, last 512 = z gate).
// ---------------------------------------------------------------------------
__global__ void dwconv_silu(const float* __restrict__ XZ,
                            const float* __restrict__ cw, const float* __restrict__ cb,
                            float* __restrict__ Uf, __bf16* __restrict__ Ub) {
    int idx = blockIdx.x * blockDim.x + threadIdx.x;
    if (idx >= 784 * 512) return;
    int d = idx % 512, n = idx / 512;
    int b = n / 49, t = n % 49;
    float acc = cb[d];
#pragma unroll
    for (int j = 0; j < 4; ++j) {
        int tt = t - 3 + j;
        if (tt >= 0) acc += XZ[((size_t)(b * 49 + tt)) * 1024 + d] * cw[d * 4 + j];
    }
    float u = acc / (1.f + __expf(-acc));   // silu
    Uf[idx] = u;
    Ub[idx] = (__bf16)u;
}

// dt = softplus(dt_in @ dt_w.T + dt_b); dt_in = xdbl[:, :16]
__global__ void dtproj_softplus(const float* __restrict__ xdbl,
                                const float* __restrict__ dtw, const float* __restrict__ dtb,
                                float* __restrict__ dt) {
    int idx = blockIdx.x * blockDim.x + threadIdx.x;
    if (idx >= 784 * 512) return;
    int d = idx % 512, n = idx / 512;
    float a = dtb[d];
#pragma unroll
    for (int r = 0; r < 16; ++r) a += xdbl[(size_t)n * 48 + r] * dtw[d * 16 + r];
    dt[idx] = (a > 20.f) ? a : log1pf(__expf(a));
}

// Selective scan: one thread per (b,d) over L=49, 16 states in registers.
__global__ void selective_scan(const float* __restrict__ dt, const float* __restrict__ Uf,
                               const float* __restrict__ xdbl, const float* __restrict__ XZ,
                               const float* __restrict__ A_log, const float* __restrict__ Dv,
                               __bf16* __restrict__ Yb) {
    int idx = blockIdx.x * blockDim.x + threadIdx.x;
    if (idx >= 16 * 512) return;
    int d = idx % 512, b = idx / 512;
    float a[16], h[16];
#pragma unroll
    for (int j = 0; j < 16; ++j) { a[j] = -__expf(A_log[d * 16 + j]); h[j] = 0.f; }
    float Dd = Dv[d];
    for (int t = 0; t < 49; ++t) {
        int n = b * 49 + t;
        float dtv = dt[(size_t)n * 512 + d];
        float uv  = Uf[(size_t)n * 512 + d];
        float y = 0.f;
#pragma unroll
        for (int j = 0; j < 16; ++j) {
            float dA = __expf(dtv * a[j]);
            float Bv = xdbl[(size_t)n * 48 + 16 + j];
            h[j] = dA * h[j] + dtv * Bv * uv;
            y += h[j] * xdbl[(size_t)n * 48 + 32 + j];
        }
        y += uv * Dd;
        float z = XZ[(size_t)n * 1024 + 512 + d];
        y *= z / (1.f + __expf(-z));
        Yb[(size_t)n * 512 + d] = (__bf16)y;
    }
}

// mean over L=49 tokens
__global__ void pool_tokens(const float* __restrict__ Tf, float* __restrict__ pooled) {
    int idx = blockIdx.x * blockDim.x + threadIdx.x;
    if (idx >= 16 * 256) return;
    int d = idx % 256, b = idx / 256;
    float s = 0.f;
    for (int l = 0; l < 49; ++l) s += Tf[(size_t)(b * 49 + l) * 256 + d];
    pooled[idx] = s * (1.f / 49.f);
}

__global__ void classifier_head(const float* __restrict__ pooled,
                                const float* __restrict__ cw, const float* __restrict__ cb,
                                float* __restrict__ out) {
    int idx = blockIdx.x * blockDim.x + threadIdx.x;
    if (idx >= 16 * 1000) return;
    int c = idx % 1000, b = idx / 1000;
    float a = cb[c];
    for (int d = 0; d < 256; ++d) a += pooled[b * 256 + d] * cw[c * 256 + d];
    out[idx] = a;
}

// ---------------------------------------------------------------------------
// Host orchestration
// ---------------------------------------------------------------------------
extern "C" void kernel_launch(void* const* d_in, const int* in_sizes, int n_in,
                              void* d_out, int out_size, void* d_ws, size_t ws_size,
                              hipStream_t stream) {
    (void)in_sizes; (void)n_in; (void)out_size; (void)ws_size;

    // ---- parse inputs (insertion order of setup_inputs / _make_params) ----
    int ip = 0;
    auto nxt = [&]() { return (const float*)d_in[ip++]; };
    const float* X = nxt();                                  // x [16,3,224,224]
    const float* stem_w = nxt(); const float* stem_s = nxt(); const float* stem_b = nxt();

    struct Blk { const float *w1,*s1,*b1,*w2,*s2,*b2,*w3,*s3,*b3,*dw,*ds,*db;
                 int in_c, mid, out_c, stride; };
    Blk blk[16]; int nb = 0;
    const int cfg[4][4] = {{64,256,3,1},{128,512,4,2},{256,1024,6,2},{512,2048,3,2}};
    int in_c = 64;
    for (int st = 0; st < 4; ++st) {
        int mid = cfg[st][0], oc = cfg[st][1], n = cfg[st][2], str = cfg[st][3];
        for (int i = 0; i < n; ++i) {
            Blk& b = blk[nb++];
            b.w1=nxt(); b.s1=nxt(); b.b1=nxt(); b.w2=nxt(); b.s2=nxt(); b.b2=nxt();
            b.w3=nxt(); b.s3=nxt(); b.b3=nxt();
            if (i == 0) { b.dw=nxt(); b.ds=nxt(); b.db=nxt(); }
            else        { b.dw=b.ds=b.db=nullptr; }
            b.in_c = (i==0)?in_c:oc; b.mid=mid; b.out_c=oc; b.stride=(i==0)?str:1;
        }
        in_c = oc;
    }
    const float* patch_w = nxt(); const float* patch_b = nxt();
    struct MB { const float *in_w,*conv_w,*conv_b,*xproj_w,*dt_w,*dt_b,*A_log,*Dv,*out_w; } mp[2];
    for (int i = 0; i < 2; ++i) {
        mp[i].in_w=nxt(); mp[i].conv_w=nxt(); mp[i].conv_b=nxt(); mp[i].xproj_w=nxt();
        mp[i].dt_w=nxt(); mp[i].dt_b=nxt(); mp[i].A_log=nxt(); mp[i].Dv=nxt(); mp[i].out_w=nxt();
    }
    const float* cls_w = nxt(); const float* cls_b = nxt();

    // ---- workspace carve-out ----
    char* ws = (char*)d_ws; size_t off = 0;
    auto alloc = [&](size_t bytes) -> char* {
        char* r = ws + off; off += (bytes + 255) & ~(size_t)255; return r;
    };
    const size_t ACT = (size_t)16 * 64 * 112 * 112;          // max activation elems
    __bf16* AB[4];
    for (int i = 0; i < 4; ++i) AB[i] = (__bf16*)alloc(ACT * 2);

    auto cvt = [&](const float* src, size_t n) -> const __bf16* {
        __bf16* dst = (__bf16*)alloc(n * 2);
        cvt_f32_bf16<<<dim3((unsigned)((n + 255) / 256)), 256, 0, stream>>>(src, dst, (int)n);
        return dst;
    };

    struct BlkBf { const __bf16 *w1,*w2,*w3,*dw; } bbf[16];
    for (int i = 0; i < 16; ++i) {
        const Blk& b = blk[i];
        bbf[i].w1 = cvt(b.w1, (size_t)b.mid * b.in_c);
        bbf[i].w2 = cvt(b.w2, (size_t)b.mid * b.mid * 9);
        bbf[i].w3 = cvt(b.w3, (size_t)b.out_c * b.mid);
        bbf[i].dw = b.dw ? cvt(b.dw, (size_t)b.out_c * b.in_c) : nullptr;
    }
    const __bf16* patch_bf = cvt(patch_w, (size_t)256 * 2048);
    const __bf16 *inw_bf[2], *xpw_bf[2], *outw_bf[2];
    for (int i = 0; i < 2; ++i) {
        inw_bf[i]  = cvt(mp[i].in_w,    (size_t)1024 * 256);
        xpw_bf[i]  = cvt(mp[i].xproj_w, (size_t)48 * 512);
        outw_bf[i] = cvt(mp[i].out_w,   (size_t)256 * 512);
    }

    float*  XZ   = (float*)alloc((size_t)784 * 1024 * 4);
    float*  Uf   = (float*)alloc((size_t)784 * 512 * 4);
    __bf16* Ub   = (__bf16*)alloc((size_t)784 * 512 * 2);
    float*  xdbl = (float*)alloc((size_t)784 * 48 * 4);
    float*  dtb  = (float*)alloc((size_t)784 * 512 * 4);
    __bf16* Yb   = (__bf16*)alloc((size_t)784 * 512 * 2);
    float*  Tf   = (float*)alloc((size_t)784 * 256 * 4);
    __bf16* Tb   = (__bf16*)alloc((size_t)784 * 256 * 2);
    float*  pooled = (float*)alloc((size_t)16 * 256 * 4);

    auto conv = [&](const __bf16* Wt, const __bf16* In, __bf16* Out,
                    const float* sc, const float* bi, const __bf16* res, int relu,
                    int M, int Cin, int Hin, int Hout, int RS1, int stride, int pad) {
        int N = 16 * Hout * Hout;
        dim3 g((N + 127) / 128, (M + 127) / 128);
        conv_gemm_bf16<<<g, TB, 0, stream>>>(Wt, In, Out, sc, bi, res, relu,
                                             M, Cin, Hin, Hin, Hout, Hout,
                                             RS1, stride, pad, 16);
    };
    auto gemm = [&](float* Out, const __bf16* Wt, const __bf16* Xr, const float* bias,
                    int M, int N, int K) {
        dim3 g((N + 127) / 128, (M + 127) / 128);
        gemm_bf16<<<g, TB, 0, stream>>>(Wt, Xr, Out, bias, M, N, K);
    };

    // ---- stem + maxpool ----
    stem_conv<<<dim3((16 * 64 * 112 * 112 + 255) / 256), 256, 0, stream>>>(
        X, AB[1], stem_w, stem_s, stem_b);
    maxpool3<<<dim3((16 * 64 * 56 * 56 + 255) / 256), 256, 0, stream>>>(AB[1], AB[0]);

    // ---- ResNet stages ----
    __bf16 *cur = AB[0], *t1 = AB[1], *t2 = AB[2], *t3 = AB[3];
    int H = 56;
    for (int i = 0; i < 16; ++i) {
        const Blk& b = blk[i];
        int Ho = H / b.stride;
        conv(bbf[i].w1, cur, t1, b.s1, b.b1, nullptr, 1, b.mid,   b.in_c, H,  H,  1, 1, 0);
        conv(bbf[i].w2, t1,  t2, b.s2, b.b2, nullptr, 1, b.mid,   b.mid,  H,  Ho, 0, b.stride, 1);
        const __bf16* scb;
        if (bbf[i].dw) {
            conv(bbf[i].dw, cur, t3, b.ds, b.db, nullptr, 0, b.out_c, b.in_c, H, Ho, 1, b.stride, 0);
            scb = t3;
        } else scb = cur;
        conv(bbf[i].w3, t2, t1, b.s3, b.b3, scb, 1, b.out_c, b.mid, Ho, Ho, 1, 1, 0);
        __bf16* tmp = cur; cur = t1; t1 = tmp;   // new cur = output; recycle
        H = Ho;
    }

    // ---- patch embed (1x1 conv 2048->256, +bias) -> tokens ----
    conv(patch_bf, cur, t2, nullptr, patch_b, nullptr, 0, 256, 2048, 7, 7, 1, 1, 0);
    tokens_from_nchw<<<dim3((784 * 256 + 255) / 256), 256, 0, stream>>>(t2, Tb);

    // ---- Mamba blocks ----
    for (int i = 0; i < 2; ++i) {
        gemm(XZ, inw_bf[i], Tb, nullptr, 1024, 784, 256);
        dwconv_silu<<<dim3((784 * 512 + 255) / 256), 256, 0, stream>>>(
            XZ, mp[i].conv_w, mp[i].conv_b, Uf, Ub);
        gemm(xdbl, xpw_bf[i], Ub, nullptr, 48, 784, 512);
        dtproj_softplus<<<dim3((784 * 512 + 255) / 256), 256, 0, stream>>>(
            xdbl, mp[i].dt_w, mp[i].dt_b, dtb);
        selective_scan<<<dim3((16 * 512 + 255) / 256), 256, 0, stream>>>(
            dtb, Uf, xdbl, XZ, mp[i].A_log, mp[i].Dv, Yb);
        gemm(Tf, outw_bf[i], Yb, nullptr, 256, 784, 512);
        cvt_f32_bf16<<<dim3((784 * 256 + 255) / 256), 256, 0, stream>>>(Tf, Tb, 784 * 256);
    }

    // ---- pool + classifier ----
    pool_tokens<<<dim3((16 * 256 + 255) / 256), 256, 0, stream>>>(Tf, pooled);
    classifier_head<<<dim3((16 * 1000 + 255) / 256), 256, 0, stream>>>(
        pooled, cls_w, cls_b, (float*)d_out);
}
